// rbf_85512798863739
// MI455X (gfx1250) — matrix operand order repaired and checked by reading, not compiled, not run on hardware
//
#include <hip/hip_runtime.h>

typedef __attribute__((ext_vector_type(16))) _Float16 v16h;
typedef __attribute__((ext_vector_type(8)))  _Float16 v8h;
typedef __attribute__((ext_vector_type(8)))  float    v8f;

#define KGRID  1024
#define LCD    20
#define DIN    24          // 20 RBF + 4 hashgrid features
#define HID    64
#define HSTR   72          // f16 elems per activation row (64 + pad, multiple of 8 -> 16B rows)
#define W0STR  40          // f16 elems per W0 row (24 -> 32 zero-pad + bank pad)
#define NWAVES 4
#define PTSW   32          // points per wave
#define BLOCKT (NWAVES * 32)
#define PTSB   (NWAVES * PTSW)

__device__ __forceinline__ void lds_fence() {
  // wave-private LDS producer->consumer ordering (cross-lane within the wave)
  asm volatile("s_wait_dscnt 0" ::: "memory");
}

__device__ __forceinline__ v16h ld16h(const _Float16* p0, const _Float16* p1) {
  v8h lo = *reinterpret_cast<const v8h*>(p0);
  v8h hi = *reinterpret_cast<const v8h*>(p1);
  return __builtin_shufflevector(lo, hi, 0,1,2,3,4,5,6,7,8,9,10,11,12,13,14,15);
}

__device__ __forceinline__ void hg_lookup(const float* __restrict__ tab, int res,
                                          float x0, float x1, float& o0, float& o1) {
  const float p0 = x0 * (float)(res - 1), p1 = x1 * (float)(res - 1);
  int u = (int)floorf(p0); u = u < 0 ? 0 : (u > res - 2 ? res - 2 : u);
  int v = (int)floorf(p1); v = v < 0 ? 0 : (v > res - 2 ? res - 2 : v);
  const float fr0 = p0 - (float)u, fr1 = p1 - (float)v;
  o0 = 0.f; o1 = 0.f;
#pragma unroll
  for (int c = 0; c < 4; ++c) {
    const int di = c >> 1, dj = c & 1;
    const float wb = (di ? fr0 : 1.f - fr0) * (dj ? fr1 : 1.f - fr1);
    const float2 t = reinterpret_cast<const float2*>(tab)[(u + di) * res + (v + dj)];
    o0 += wb * t.x; o1 += wb * t.y;
  }
}

__global__ __launch_bounds__(BLOCKT) void fused_field_mlp(
    const float* __restrict__ x,
    const float* __restrict__ hg0, const float* __restrict__ hg1,
    const float* __restrict__ kc0, const float* __restrict__ ks0,
    const float* __restrict__ lc0, const float* __restrict__ lcb0,
    const float* __restrict__ W0, const float* __restrict__ b0,
    const float* __restrict__ W1, const float* __restrict__ b1,
    const float* __restrict__ W2, const float* __restrict__ b2,
    const float* __restrict__ a0p, const float* __restrict__ a1p, const float* __restrict__ a2p,
    float* __restrict__ out, int npts)
{
  __shared__ __align__(16) _Float16 sW0[HID * W0STR];        //  5.0 KB
  __shared__ __align__(16) _Float16 sW1[HID * HSTR];         //  9.0 KB
  __shared__ __align__(16) _Float16 sW2[16 * HSTR];          //  2.25 KB
  __shared__ __align__(16) _Float16 sAct[NWAVES][2][PTSW * HSTR]; // 36 KB

  const int tid = threadIdx.x;

  // ---- stage W: convert weights to f16 in LDS (zero K-padding) ----
  for (int i = tid; i < HID * W0STR; i += BLOCKT) {
    const int r = i / W0STR, c = i - r * W0STR;
    sW0[i] = (c < DIN) ? (_Float16)W0[r * DIN + c] : (_Float16)0.f;
  }
  for (int i = tid; i < HID * HSTR; i += BLOCKT) {
    const int r = i / HSTR, c = i - r * HSTR;
    sW1[i] = (c < HID) ? (_Float16)W1[r * HID + c] : (_Float16)0.f;
  }
  for (int i = tid; i < 16 * HSTR; i += BLOCKT) {
    const int r = i / HSTR, c = i - r * HSTR;
    sW2[i] = (r < 3 && c < HID) ? (_Float16)W2[r * HID + c] : (_Float16)0.f;
  }
  __syncthreads();

  const int wave = tid >> 5;
  const int lane = tid & 31;
  const int base = blockIdx.x * PTSB + wave * PTSW;
  _Float16* bufA = &sAct[wave][0][0];   // features, later h2
  _Float16* bufB = &sAct[wave][1][0];   // h1

  const float a0s = a0p[0], a1s = a1p[0], a2s = a2p[0];

  // ---- stage A: per-lane feature build (fp32 gathers) ----
  {
    int p = base + lane; if (p > npts - 1) p = npts - 1;
    const float2 xv = reinterpret_cast<const float2*>(x)[p];
    const float x0 = xv.x, x1 = xv.y;
    float f[DIN];

    // RBF: cell corners on the regular K x K grid
    const float g0 = x0 * (float)(KGRID - 1);
    const float g1 = x1 * (float)(KGRID - 1);
    int i0 = (int)floorf(g0); i0 = i0 < 0 ? 0 : (i0 > KGRID - 2 ? KGRID - 2 : i0);
    int j0 = (int)floorf(g1); j0 = j0 < 0 ? 0 : (j0 > KGRID - 2 ? KGRID - 2 : j0);
    float w[4]; int idx[4]; float wsum = 0.f;
#pragma unroll
    for (int c = 0; c < 4; ++c) {
      const int id = (i0 + (c >> 1)) * KGRID + (j0 + (c & 1));
      idx[c] = id;
      const float2 kc = reinterpret_cast<const float2*>(kc0)[id];
      const float ks = ks0[id];
      const float t0 = fmaxf(1.f - fabsf(x0 - kc.x) * ks, 0.f);
      const float t1 = fmaxf(1.f - fabsf(x1 - kc.y) * ks, 0.f);
      w[c] = t0 * t1;  wsum += w[c];
    }
    const float inv = 1.f / (wsum + 1e-8f);
#pragma unroll
    for (int d = 0; d < LCD; ++d) f[d] = 0.f;
#pragma unroll
    for (int c = 0; c < 4; ++c) {
      const float wc = w[c] * inv;
      const float4* row = reinterpret_cast<const float4*>(lc0 + (size_t)idx[c] * LCD);
#pragma unroll
      for (int q = 0; q < 5; ++q) {
        const float4 t = row[q];
        f[4 * q + 0] += wc * t.x; f[4 * q + 1] += wc * t.y;
        f[4 * q + 2] += wc * t.z; f[4 * q + 3] += wc * t.w;
      }
    }
    hg_lookup(hg0, 16,   x0, x1, f[20], f[21]);
    hg_lookup(hg1, 2048, x0, x1, f[22], f[23]);
#pragma unroll
    for (int d = 0; d < DIN; ++d) f[d] += lcb0[d];   // uniform addrs -> s_loads

    // pack 24 features (+8 zero pad) into 4 x b128 LDS stores
    v8h h0, h1, h2, h3;
#pragma unroll
    for (int i = 0; i < 8; ++i) {
      h0[i] = (_Float16)f[i];
      h1[i] = (_Float16)f[8 + i];
      h2[i] = (_Float16)f[16 + i];
      h3[i] = (_Float16)0.f;
    }
    v8h* frow = reinterpret_cast<v8h*>(bufA + lane * HSTR);
    frow[0] = h0;  frow[1] = h1;  frow[2] = h2;  frow[3] = h3;
  }
  lds_fence();

  const int n = lane & 15;   // column (and A-row) index within tile
  const int g = lane >> 4;   // K half-group

  // ---- layer 0: 24(->32) -> 64, one 16x16x32 WMMA per N-tile ----
#pragma unroll
  for (int mt = 0; mt < 2; ++mt) {
    const _Float16* arow = bufA + (mt * 16 + n) * HSTR;
    const v16h a = ld16h(arow + 8 * g, arow + 16 + 8 * g);
#pragma unroll
    for (int nt = 0; nt < 4; ++nt) {
      const _Float16* brow = sW0 + (nt * 16 + n) * W0STR;
      const v16h b = ld16h(brow + 8 * g, brow + 16 + 8 * g);
      const float bv = b0[nt * 16 + n];
      v8f c = { bv, bv, bv, bv, bv, bv, bv, bv };
      c = __builtin_amdgcn_wmma_f32_16x16x32_f16(false, a, false, b, (short)0, c, false, false);
#pragma unroll
      for (int r = 0; r < 8; ++r) {
        const float v = fmaxf(a0s * c[r], 0.f);
        bufB[(mt * 16 + r + 8 * g) * HSTR + nt * 16 + n] = (_Float16)v;
      }
    }
  }
  lds_fence();

  // ---- layer 1: 64 -> 64, two K-steps per N-tile ----
#pragma unroll
  for (int mt = 0; mt < 2; ++mt) {
    const _Float16* arow = bufB + (mt * 16 + n) * HSTR;
    const v16h aA = ld16h(arow + 8 * g,      arow + 16 + 8 * g);
    const v16h aB = ld16h(arow + 32 + 8 * g, arow + 48 + 8 * g);
#pragma unroll
    for (int nt = 0; nt < 4; ++nt) {
      const _Float16* brow = sW1 + (nt * 16 + n) * HSTR;
      const v16h bA = ld16h(brow + 8 * g,      brow + 16 + 8 * g);
      const v16h bB = ld16h(brow + 32 + 8 * g, brow + 48 + 8 * g);
      const float bv = b1[nt * 16 + n];
      v8f c = { bv, bv, bv, bv, bv, bv, bv, bv };
      c = __builtin_amdgcn_wmma_f32_16x16x32_f16(false, aA, false, bA, (short)0, c, false, false);
      c = __builtin_amdgcn_wmma_f32_16x16x32_f16(false, aB, false, bB, (short)0, c, false, false);
#pragma unroll
      for (int r = 0; r < 8; ++r) {
        const float v = fmaxf(a1s * c[r], 0.f);
        bufA[(mt * 16 + r + 8 * g) * HSTR + nt * 16 + n] = (_Float16)v;   // feat is dead
      }
    }
  }
  lds_fence();

  // ---- layer 2: 64 -> 3 (one N-tile, columns 0..2 valid) ----
#pragma unroll
  for (int mt = 0; mt < 2; ++mt) {
    const _Float16* arow = bufA + (mt * 16 + n) * HSTR;
    const v16h aA = ld16h(arow + 8 * g,      arow + 16 + 8 * g);
    const v16h aB = ld16h(arow + 32 + 8 * g, arow + 48 + 8 * g);
    const _Float16* brow = sW2 + n * HSTR;
    const v16h bA = ld16h(brow + 8 * g,      brow + 16 + 8 * g);
    const v16h bB = ld16h(brow + 32 + 8 * g, brow + 48 + 8 * g);
    const float bv = (n < 3) ? b2[n] : 0.f;
    v8f c = { bv, bv, bv, bv, bv, bv, bv, bv };
    c = __builtin_amdgcn_wmma_f32_16x16x32_f16(false, aA, false, bA, (short)0, c, false, false);
    c = __builtin_amdgcn_wmma_f32_16x16x32_f16(false, aB, false, bB, (short)0, c, false, false);
    if (n < 3) {
#pragma unroll
      for (int r = 0; r < 8; ++r) {
        const int p = base + mt * 16 + r + 8 * g;
        if (p < npts) out[(size_t)p * 3 + n] = a2s * c[r];
      }
    }
  }
}

extern "C" void kernel_launch(void* const* d_in, const int* in_sizes, int n_in,
                              void* d_out, int out_size, void* d_ws, size_t ws_size,
                              hipStream_t stream) {
  const float* x    = (const float*)d_in[0];
  const float* hg0  = (const float*)d_in[1];
  const float* hg1  = (const float*)d_in[2];
  const float* kc0  = (const float*)d_in[3];
  const float* ks0  = (const float*)d_in[4];
  const float* lc0  = (const float*)d_in[5];
  const float* lcb0 = (const float*)d_in[6];
  const float* W0   = (const float*)d_in[7];
  const float* b0   = (const float*)d_in[8];
  const float* W1   = (const float*)d_in[9];
  const float* b1   = (const float*)d_in[10];
  const float* W2   = (const float*)d_in[11];
  const float* b2   = (const float*)d_in[12];
  const float* a0   = (const float*)d_in[13];
  const float* a1   = (const float*)d_in[14];
  const float* a2   = (const float*)d_in[15];
  float* out = (float*)d_out;

  const int npts = in_sizes[0] / 2;           // x is [N,2]
  const int blocks = (npts + PTSB - 1) / PTSB;
  fused_field_mlp<<<blocks, BLOCKT, 0, stream>>>(
      x, hg0, hg1, kc0, ks0, lc0, lcb0, W0, b0, W1, b1, W2, b2, a0, a1, a2, out, npts);
}
